// Minerva_with_encoding_37211596652566
// MI455X (gfx1250) — matrix operand order repaired and checked by reading, not compiled, not run on hardware
//
#include <hip/hip_runtime.h>
#include <hip/hip_bf16.h>

typedef _Float16 hvec8  __attribute__((ext_vector_type(8)));
typedef _Float16 hvec16 __attribute__((ext_vector_type(16)));
typedef float    f32x8  __attribute__((ext_vector_type(8)));

#define N_Q    4096
#define N_D    8192
#define D_IN   768
#define REP    768
#define R_DIMS 4
#define N_ENC  129
#define SACT_LD 136   // 128 + 8 halves pad: row-to-row shift of 4 banks

// ---------------------------------------------------------------------------
// Fragment loaders for v_wmma_f32_16x16x32_f16 (wave32).
// A (16x32 f16, MxK): lane L holds row m=L&15; halves 0..7 = K[kb..kb+7],
// halves 8..15 = K[kb+16..kb+23], kb=(L<16)?0:8.  Two 16B contiguous loads.
// B (32x16 f16, KxN): lane L holds col n=L&15; halves 0..7 = K[ko..ko+7],
// halves 8..15 = K[ko+16..ko+23], ko=(L<16)?0:8 — same order as A.
// ---------------------------------------------------------------------------
__device__ __forceinline__ hvec16 load_frag_a(const _Float16* __restrict__ p,
                                              int ld, int row0, int k0) {
    const int L  = threadIdx.x & 31;
    const int m  = L & 15;
    const int kb = (L < 16) ? 0 : 8;
    const _Float16* base = p + (size_t)(row0 + m) * ld + (k0 + kb);
    hvec8 lo = *(const hvec8*)(base);
    hvec8 hi = *(const hvec8*)(base + 16);
    hvec16 r;
#pragma unroll
    for (int i = 0; i < 8; ++i) { r[i] = lo[i]; r[i + 8] = hi[i]; }
    return r;
}

__device__ __forceinline__ hvec16 load_frag_b(const _Float16* __restrict__ q,
                                              int ld, int col0, int k0) {
    const int L  = threadIdx.x & 31;
    const int n  = L & 15;
    const int ko = (L < 16) ? 0 : 8;
    const _Float16* base = q + (size_t)(col0 + n) * ld + (k0 + ko);
    hvec8 lo = *(const hvec8*)(base);
    hvec8 hi = *(const hvec8*)(base + 16);
    hvec16 r;
#pragma unroll
    for (int i = 0; i < 8; ++i) { r[i] = lo[i]; r[i + 8] = hi[i]; }
    return r;
}

// ---------------------------------------------------------------------------
// fp32 -> fp16 conversion (vectorized x4)
// ---------------------------------------------------------------------------
__global__ void cvt_f16_kernel(const float* __restrict__ src,
                               _Float16* __restrict__ dst, int n4) {
    int i = blockIdx.x * blockDim.x + threadIdx.x;
    if (i >= n4) return;
    float4 v = ((const float4*)src)[i];
    _Float16* d = dst + (size_t)i * 4;
    d[0] = (_Float16)v.x; d[1] = (_Float16)v.y;
    d[2] = (_Float16)v.z; d[3] = (_Float16)v.w;
}

// ---------------------------------------------------------------------------
// Nearest-encoding lookup -> K-major f16 operand RencT[16][N_D]
// (rows 0..3 = pos_encoding[argmin].T, rows 4..15 = 0 for WMMA N-padding)
// ---------------------------------------------------------------------------
__global__ void renc_kernel(const float* __restrict__ R,
                            const float* __restrict__ enc,
                            const float* __restrict__ pos,
                            _Float16* __restrict__ RencT) {
    int m = blockIdx.x * blockDim.x + threadIdx.x;
    if (m >= N_D) return;
    float r = R[m];
    int best = 0;
    float bd = fabsf(r - enc[0]);
    for (int j = 1; j < N_ENC; ++j) {
        float d = fabsf(r - enc[j]);
        if (d < bd) { bd = d; best = j; }   // strict < => first-min (argmin)
    }
    const float* p = pos + (size_t)best * R_DIMS;
#pragma unroll
    for (int rd = 0; rd < 16; ++rd)
        RencT[(size_t)rd * N_D + m] = (rd < R_DIMS) ? (_Float16)p[rd] : (_Float16)0.f;
}

// ---------------------------------------------------------------------------
// Projection GEMM:  out[n,r] = f16( sum_k A[n,k]*W[r,k] + bias[r] )
// Workgroup tile 128x128, 8 waves (2x4), wave tile 64x32 (4x2 fragments).
// ---------------------------------------------------------------------------
__global__ __launch_bounds__(256)
void proj_kernel(const _Float16* __restrict__ A, const _Float16* __restrict__ W,
                 const float* __restrict__ bias, _Float16* __restrict__ out) {
    const int lane  = threadIdx.x & 31;
    const int wave  = threadIdx.x >> 5;
    const int tileM = blockIdx.x * 128;
    const int tileN = blockIdx.y * 128;
    const int wm    = (wave >> 2) * 64;
    const int wn    = (wave & 3) * 32;

    f32x8 acc[4][2] = {};
    for (int k0 = 0; k0 < D_IN; k0 += 32) {
        hvec16 af[4], bf[2];
#pragma unroll
        for (int i = 0; i < 4; ++i) af[i] = load_frag_a(A, D_IN, tileM + wm + 16 * i, k0);
#pragma unroll
        for (int j = 0; j < 2; ++j) bf[j] = load_frag_b(W, D_IN, tileN + wn + 16 * j, k0);
#pragma unroll
        for (int i = 0; i < 4; ++i)
#pragma unroll
            for (int j = 0; j < 2; ++j)
                acc[i][j] = __builtin_amdgcn_wmma_f32_16x16x32_f16(
                    false, af[i], false, bf[j], (short)0, acc[i][j], false, false);
    }

    const int mL = (lane < 16) ? 0 : 8;
    const int nL = lane & 15;
#pragma unroll
    for (int j = 0; j < 2; ++j) {
        const int col = tileN + wn + 16 * j + nL;
        const float b = bias[col];
#pragma unroll
        for (int i = 0; i < 4; ++i) {
#pragma unroll
            for (int r = 0; r < 8; ++r) {
                const int row = tileM + wm + 16 * i + r + mL;
                out[(size_t)row * REP + col] = (_Float16)(acc[i][j][r] + b);
            }
        }
    }
}

// ---------------------------------------------------------------------------
// Fused similarity GEMM + power-sign activation + echo GEMM.
// Workgroup: 128-query tile x 1024-doc slab (8 inner 128-doc tiles).
// Per tile: a_tile via WMMA -> act=v*|v| staged as f16 in LDS ->
// echo accumulated by a second WMMA against RencT (no atomics; each
// workgroup writes an exclusive echo_part[slab][query][rd] region).
// ---------------------------------------------------------------------------
__global__ __launch_bounds__(256)
void score_echo_kernel(const _Float16* __restrict__ Xw,
                       const _Float16* __restrict__ Dw,
                       const _Float16* __restrict__ RencT,
                       float* __restrict__ echo_part) {
    __shared__ _Float16 sAct[128][SACT_LD];
    const int lane  = threadIdx.x & 31;
    const int wave  = threadIdx.x >> 5;
    const int tileM = blockIdx.x * 128;
    const int slab  = blockIdx.y;          // 0..7, each owns 1024 docs
    const int wm    = (wave >> 2) * 64;
    const int wn    = (wave & 3) * 32;
    const int mL    = (lane < 16) ? 0 : 8;
    const int nL    = lane & 15;

    f32x8 accE = {};                       // echo accumulator, carried over slab

    for (int ct = 0; ct < 8; ++ct) {
        const int tileN = slab * 1024 + ct * 128;
        // ---- main GEMM: 64x32 wave tile over K=768 --------------------
        f32x8 acc[4][2] = {};
        for (int k0 = 0; k0 < REP; k0 += 32) {
            hvec16 af[4], bf[2];
#pragma unroll
            for (int i = 0; i < 4; ++i) af[i] = load_frag_a(Xw, REP, tileM + wm + 16 * i, k0);
#pragma unroll
            for (int j = 0; j < 2; ++j) bf[j] = load_frag_b(Dw, REP, tileN + wn + 16 * j, k0);
#pragma unroll
            for (int i = 0; i < 4; ++i)
#pragma unroll
                for (int j = 0; j < 2; ++j)
                    acc[i][j] = __builtin_amdgcn_wmma_f32_16x16x32_f16(
                        false, af[i], false, bf[j], (short)0, acc[i][j], false, false);
        }
        // ---- activation: act = v*|v|, stage f16 tile in LDS -----------
#pragma unroll
        for (int j = 0; j < 2; ++j) {
            const int col = wn + 16 * j + nL;
#pragma unroll
            for (int i = 0; i < 4; ++i) {
#pragma unroll
                for (int r = 0; r < 8; ++r) {
                    const float v = acc[i][j][r];
                    sAct[wm + 16 * i + r + mL][col] = (_Float16)(v * fabsf(v));
                }
            }
        }
        __syncthreads();
        // ---- echo GEMM: rows[16*wave..+15] x RencT over K=128 ---------
#pragma unroll
        for (int kf = 0; kf < 4; ++kf) {
            hvec16 a;
            {
                const int kb = (lane < 16) ? 0 : 8;
                const _Float16* base = &sAct[16 * wave + nL][32 * kf + kb];
                hvec8 lo = *(const hvec8*)(base);
                hvec8 hi = *(const hvec8*)(base + 16);
#pragma unroll
                for (int i = 0; i < 8; ++i) { a[i] = lo[i]; a[i + 8] = hi[i]; }
            }
            hvec16 b = load_frag_b(RencT, N_D, 0, tileN + 32 * kf);
            accE = __builtin_amdgcn_wmma_f32_16x16x32_f16(
                false, a, false, b, (short)0, accE, false, false);
        }
        __syncthreads();   // protect sAct before next tile's stores
    }

    // ---- write exclusive slab partial: lane col nL = rd (only 0..3) ----
    if (nL < R_DIMS) {
#pragma unroll
        for (int r = 0; r < 8; ++r) {
            const int row = tileM + 16 * wave + r + mL;
            echo_part[((size_t)slab * N_Q + row) * R_DIMS + nL] = accE[r];
        }
    }
}

// ---------------------------------------------------------------------------
// Final: out[n] = sum_s echo_part[s][n] . We_w + We_b
// ---------------------------------------------------------------------------
__global__ void out_kernel(const float* __restrict__ echo_part,
                           const float* __restrict__ We_w,
                           const float* __restrict__ We_b,
                           float* __restrict__ out) {
    int n = blockIdx.x * blockDim.x + threadIdx.x;
    if (n >= N_Q) return;
    float e0 = 0.f, e1 = 0.f, e2 = 0.f, e3 = 0.f;
    for (int s = 0; s < 8; ++s) {
        const float* p = echo_part + ((size_t)s * N_Q + n) * R_DIMS;
        e0 += p[0]; e1 += p[1]; e2 += p[2]; e3 += p[3];
    }
    out[n] = e0 * We_w[0] + e1 * We_w[1] + e2 * We_w[2] + e3 * We_w[3] + We_b[0];
}

// ---------------------------------------------------------------------------
extern "C" void kernel_launch(void* const* d_in, const int* in_sizes, int n_in,
                              void* d_out, int out_size, void* d_ws, size_t ws_size,
                              hipStream_t stream) {
    const float* X      = (const float*)d_in[0];   // [4096,768]
    const float* D      = (const float*)d_in[1];   // [8192,768]
    const float* R      = (const float*)d_in[2];   // [8192,1]
    const float* Wx_w   = (const float*)d_in[3];   // [768,768]
    const float* Wx_b   = (const float*)d_in[4];   // [768]
    const float* Wd_w   = (const float*)d_in[5];   // [768,768]
    const float* Wd_b   = (const float*)d_in[6];   // [768]
    const float* We_w   = (const float*)d_in[7];   // [1,4]
    const float* We_b   = (const float*)d_in[8];   // [1]
    const float* encIds = (const float*)d_in[9];   // [129]
    const float* posEnc = (const float*)d_in[10];  // [129,4]
    float* out = (float*)d_out;

    // workspace carve-up
    char* ws = (char*)d_ws;
    _Float16* Xh    = (_Float16*)ws;  ws += (size_t)N_Q * D_IN * 2;
    _Float16* Dh    = (_Float16*)ws;  ws += (size_t)N_D * D_IN * 2;
    _Float16* Wxh   = (_Float16*)ws;  ws += (size_t)REP * D_IN * 2;
    _Float16* Wdh   = (_Float16*)ws;  ws += (size_t)REP * D_IN * 2;
    _Float16* Xwh   = (_Float16*)ws;  ws += (size_t)N_Q * REP * 2;
    _Float16* Dwh   = (_Float16*)ws;  ws += (size_t)N_D * REP * 2;
    _Float16* RencT = (_Float16*)ws;  ws += (size_t)16 * N_D * 2;
    float*    echo  = (float*)ws;     ws += (size_t)8 * N_Q * R_DIMS * 4;

    const int blk = 256;
    // 1) fp32 -> fp16 conversions
    {
        int n4;
        n4 = N_Q * D_IN / 4; cvt_f16_kernel<<<(n4 + blk - 1) / blk, blk, 0, stream>>>(X, Xh, n4);
        n4 = N_D * D_IN / 4; cvt_f16_kernel<<<(n4 + blk - 1) / blk, blk, 0, stream>>>(D, Dh, n4);
        n4 = REP * D_IN / 4; cvt_f16_kernel<<<(n4 + blk - 1) / blk, blk, 0, stream>>>(Wx_w, Wxh, n4);
        n4 = REP * D_IN / 4; cvt_f16_kernel<<<(n4 + blk - 1) / blk, blk, 0, stream>>>(Wd_w, Wdh, n4);
    }
    // 2) nearest-encoding gather -> K-major f16 RencT
    renc_kernel<<<(N_D + blk - 1) / blk, blk, 0, stream>>>(R, encIds, posEnc, RencT);
    // 3) projections (WMMA)
    proj_kernel<<<dim3(N_Q / 128, REP / 128), blk, 0, stream>>>(Xh, Wxh, Wx_b, Xwh);
    proj_kernel<<<dim3(N_D / 128, REP / 128), blk, 0, stream>>>(Dh, Wdh, Wd_b, Dwh);
    // 4) fused similarity GEMM + activation + echo GEMM (WMMA)
    score_echo_kernel<<<dim3(N_Q / 128, 8), blk, 0, stream>>>(Xwh, Dwh, RencT, echo);
    // 5) final linear
    out_kernel<<<(N_Q + blk - 1) / blk, blk, 0, stream>>>(echo, We_w, We_b, out);
    (void)in_sizes; (void)n_in; (void)out_size; (void)ws_size;
}
